// WeatherCodebook_10917806866908
// MI455X (gfx1250) — matrix operand order repaired and checked by reading, not compiled, not run on hardware
//
#include <hip/hip_runtime.h>
#include <hip/hip_bf16.h>
#include <math.h>
#include <stdint.h>

// Problem constants (from reference): B=16, C=D=256, H=W=64, K=1024
#define NBATCH 16
#define CCH    256
#define HW     4096
#define NPOS   (NBATCH * HW)      // 65536 positions
#define KCB    1024               // codebook entries
#define POS_PER_BLK 128
#define NBLK   (NPOS / POS_PER_BLK)  // 512 blocks

typedef __attribute__((ext_vector_type(2)))  float  v2f;
typedef __attribute__((ext_vector_type(8)))  float  v8f;
typedef __attribute__((ext_vector_type(8)))  __bf16 v8bf;
typedef __attribute__((ext_vector_type(16))) __bf16 v16bf;

// LDS layout (dynamic, ~111 KB of 320 KB/WGP -> 2 WGs resident):
//   fb   : bf16 features [128][FB_STRIDE]          (phases 1-2)
//   stg0/1: f32 tile stage buffers [16][WT_STRIDE] (async double-buffer,
//           W tiles in phase 1, codebook tiles in phase 2)
//   et   : bf16 codebook tile [16][ET_STRIDE]      (phase 2)
//   invn : f32[128]; bsh : int[128]; pn : f32[256] (||e||^2 partials)
#define FB_STRIDE 264   // 256 + 8 pad -> 528B row, 16B aligned
#define WT_STRIDE 260   // 1040B row, 8B aligned for v2f
#define ET_STRIDE 264
#define FB_BYTES  (POS_PER_BLK * FB_STRIDE * 2)  // 67584
#define STG_BYTES (16 * WT_STRIDE * 4)           // 16640
#define ET_BYTES  (16 * ET_STRIDE * 2)           // 8448
#define OFF_STG0  (FB_BYTES)
#define OFF_STG1  (FB_BYTES + STG_BYTES)
#define OFF_ET    (FB_BYTES + 2 * STG_BYTES)
#define OFF_INVN  (OFF_ET + ET_BYTES)
#define OFF_BSH   (OFF_INVN + 512)
#define OFF_PN    (OFF_BSH + 512)
#define SMEM_BYTES (OFF_PN + 1024)               // 111360

// ---- CDNA5 async global->LDS copy (ASYNCcnt-tracked) -----------------------
__device__ __forceinline__ void async_load_b128(uint32_t lds_addr,
                                                const float* gaddr) {
    asm volatile("global_load_async_to_lds_b128 %0, %1, off"
                 :: "v"(lds_addr), "v"((unsigned long long)(uintptr_t)gaddr)
                 : "memory");
}

#if __has_builtin(__builtin_amdgcn_s_wait_asynccnt)
#define WAIT_ASYNC(n) __builtin_amdgcn_s_wait_asynccnt(n)
#else
#define WAIT_ASYNC(n) asm volatile("s_wait_asynccnt %0" :: "i"(n) : "memory")
#endif

// Stage one 16x256 f32 tile (row-major in global) into a stride-260 LDS
// buffer. 32 async b128 issues block-wide = 4 per wave; all 32 lanes of an
// issue read 512 consecutive bytes (coalesced).
__device__ __forceinline__ void stage_tile_async(const float* __restrict__ gsrc,
                                                 float* sdst, int tid) {
    const int l = tid & 31;
    const int w = tid >> 5;
    #pragma unroll
    for (int i = 0; i < 4; ++i) {
        const int q = w * 4 + i;              // chunk 0..31 (half-row each)
        const int r = q >> 1;
        const int c = (q & 1) * 128 + l * 4;
        const uint32_t lds = (uint32_t)(uintptr_t)(sdst + r * WT_STRIDE + c);
        async_load_b128(lds, gsrc + r * CCH + c);
    }
}

__global__ __launch_bounds__(256, 1)
void vq_fused_kernel(const float* __restrict__ x,      // [16,256,64,64] NCHW
                     const float* __restrict__ conv_w, // [256,256] (o,c)
                     const float* __restrict__ conv_b, // [256]
                     const float* __restrict__ cbook,  // [1024,256]
                     float* __restrict__ out)          // [16,256,64,64] NCHW
{
    extern __shared__ char smem[];
    __bf16* fb   = (__bf16*)smem;
    float*  stg0 = (float*)(smem + OFF_STG0);
    float*  stg1 = (float*)(smem + OFF_STG1);
    __bf16* et   = (__bf16*)(smem + OFF_ET);
    float*  invn = (float*)(smem + OFF_INVN);
    int*    bsh  = (int*)  (smem + OFF_BSH);
    float*  pn   = (float*)(smem + OFF_PN);

    const int tid   = threadIdx.x;
    const int lane  = tid & 31;
    const int wave  = tid >> 5;          // 8 waves, 16 positions each
    const int l16   = lane & 15;
    const int lhalf = lane >> 4;         // 0: lanes 0-15, 1: lanes 16-31

    const int n0   = blockIdx.x * POS_PER_BLK;
    const int bidx = n0 >> 12;           // batch (128 | 4096)
    const int hw0  = n0 & (HW - 1);
    const int pm0  = wave * 16;          // wave's position tile inside block
    const int hwp  = hw0 + pm0 + l16;    // this lane's column position
    const float* xb = x + ((size_t)bidx << 20);   // b * 256*4096

    // ---------------- Phase 1: 1x1 conv via V_WMMA_F32_16X16X4_F32 ---------
    // A = W tile (M=16 outs x K), B = x (K x N=16 positions), D[m=o][n=pos].
    // W tiles double-buffered through LDS with async-to-LDS copies.
    stage_tile_async(conv_w, stg0, tid);              // tile 0 in flight
    float ssq = 0.0f;
    for (int ot = 0; ot < 16; ++ot) {
        const float* wtc = (ot & 1) ? stg1 : stg0;
        if (ot + 1 < 16) {
            stage_tile_async(conv_w + (size_t)(ot + 1) * 16 * CCH,
                             ((ot + 1) & 1) ? stg1 : stg0, tid);
            WAIT_ASYNC(4);               // newest tile may stay in flight
        } else {
            WAIT_ASYNC(0);
        }
        __syncthreads();                 // tile ot landed for all waves

        v8f acc = {};
        const float* wrow = wtc + l16 * WT_STRIDE;  // A row: out = ot*16 + l16
        #pragma unroll 8
        for (int k = 0; k < CCH; k += 4) {
            const int kk = k + 2 * lhalf;
            // A frag: lanes0-15 K={k,k+1}, lanes16-31 K={k+2,k+3}
            v2f a = *(const v2f*)(wrow + kk);
            // B frag: same K split, column n = l16 -> x[b][kk][hwp]
            v2f bv;
            bv[0] = xb[((size_t)kk << 12) + hwp];
            bv[1] = xb[((size_t)(kk + 1) << 12) + hwp];
            acc = __builtin_amdgcn_wmma_f32_16x16x4_f32(
                false, a, false, bv, (short)0, acc, false, false);
        }
        // D tile: lane holds col n=l16 (pos pm0+l16), rows m = v + 8*lhalf.
        v8bf pk;
        #pragma unroll
        for (int v = 0; v < 8; ++v) {
            const int o = ot * 16 + 8 * lhalf + v;
            float f = acc[v] + conv_b[o];
            ssq += f * f;
            pk[v] = (__bf16)f;
        }
        *(v8bf*)(fb + (size_t)(pm0 + l16) * FB_STRIDE + ot * 16 + 8 * lhalf) = pk;
        __syncthreads();                 // buffer reused by issue at ot+1
    }

    // row sum-of-squares: combine the two half-wave partials, 1/max(||f||,eps)
    ssq += __shfl_xor(ssq, 16, 32);
    if (lhalf == 0) invn[pm0 + l16] = 1.0f / fmaxf(sqrtf(ssq), 1e-12f);
    __syncthreads();

    // ---------------- Phase 1.5: normalize bf16 features in LDS ------------
    {
        const int p  = tid >> 1;
        const int c0 = (tid & 1) * 128;
        const float inv = invn[p];
        #pragma unroll
        for (int cg = 0; cg < 128; cg += 8) {
            v8bf* ptr = (v8bf*)(fb + (size_t)p * FB_STRIDE + c0 + cg);
            v8bf vv = *ptr;
            #pragma unroll
            for (int i = 0; i < 8; ++i) vv[i] = (__bf16)((float)vv[i] * inv);
            *ptr = vv;
        }
    }

    // ---------------- Phase 2: codebook scan via V_WMMA_F32_16X16X32_BF16 --
    // argmin ||f-e||^2  ==  argmax (f.e - 0.5*||e||^2).  A = features
    // (M=pos, K=dim), B = codebook tile (K x N=16 codes), D[m=pos][n=code].
    // Codebook tiles stream through the same async double-buffer; f32->bf16
    // conversion + ||e||^2 partials happen LDS->LDS after the wait.
    float best[8];
    int   bcode[8];
    #pragma unroll
    for (int v = 0; v < 8; ++v) { best[v] = -3.4e38f; bcode[v] = 0; }

    const __bf16* arow = fb + (size_t)(pm0 + l16) * FB_STRIDE;

    stage_tile_async(cbook, stg0, tid);               // tile 0 in flight
    for (int j = 0; j < KCB / 16; ++j) {
        const float* sc = (j & 1) ? stg1 : stg0;
        if (j + 1 < KCB / 16) {
            stage_tile_async(cbook + (size_t)(j + 1) * 16 * CCH,
                             ((j + 1) & 1) ? stg1 : stg0, tid);
            WAIT_ASYNC(4);
        } else {
            WAIT_ASYNC(0);
        }
        __syncthreads();                 // tile j landed; et free (prev done)

        {   // convert f32 stage -> bf16 tile, ||e||^2 partials (no atomics)
            const int r  = tid >> 4;
            const int c0 = (tid & 15) * 16;
            const float* sr = sc + r * WT_STRIDE + c0;
            __bf16* de = et + r * ET_STRIDE + c0;
            float ps = 0.0f;
            #pragma unroll
            for (int i = 0; i < 16; ++i) {
                const float v = sr[i];
                ps += v * v;
                de[i] = (__bf16)v;
            }
            pn[tid] = ps;                // pn[r*16 + (c0/16)]
        }
        __syncthreads();                 // et + pn ready (also guards stg reuse)

        v8f acc = {};
        const __bf16* brow = et + l16 * ET_STRIDE;  // column n=l16 = code row
        #pragma unroll
        for (int k0 = 0; k0 < CCH; k0 += 32) {
            // A frag (16x32 bf16): v0-3 = K[k0 + 8*lhalf ..+7], v4-7 = +16
            v8bf alo = *(const v8bf*)(arow + k0 + 8 * lhalf);
            v8bf ahi = *(const v8bf*)(arow + k0 + 16 + 8 * lhalf);
            v16bf af = __builtin_shufflevector(alo, ahi,
                0,1,2,3,4,5,6,7,8,9,10,11,12,13,14,15);
            // B frag (32x16 bf16): v0-3 = K[k0 + 8*lhalf ..+7], v4-7 = +16
            v8bf blo = *(const v8bf*)(brow + k0 + 8 * lhalf);
            v8bf bhi = *(const v8bf*)(brow + k0 + 16 + 8 * lhalf);
            v16bf bfv = __builtin_shufflevector(blo, bhi,
                0,1,2,3,4,5,6,7,8,9,10,11,12,13,14,15);
            acc = __builtin_amdgcn_wmma_f32_16x16x32_bf16(
                false, af, false, bfv, (short)0, acc, false, false);
        }
        float cn = 0.0f;                 // ||e||^2 for code column l16
        #pragma unroll
        for (int i = 0; i < 16; ++i) cn += pn[l16 * 16 + i];
        const float halfn = 0.5f * cn;
        const int   code  = j * 16 + l16;
        #pragma unroll
        for (int v = 0; v < 8; ++v) {
            const float s = acc[v] - halfn;
            if (s > best[v]) { best[v] = s; bcode[v] = code; }
        }
    }

    // reduce argmax across the 16 lanes holding each row's columns
    #pragma unroll
    for (int v = 0; v < 8; ++v) {
        float s = best[v];
        int   c = bcode[v];
        #pragma unroll
        for (int m = 8; m >= 1; m >>= 1) {
            float s2 = __shfl_xor(s, m, 32);
            int   c2 = __shfl_xor(c, m, 32);
            if (s2 > s || (s2 == s && c2 < c)) { s = s2; c = c2; }
        }
        if (l16 == 0) bsh[pm0 + 8 * lhalf + v] = c;  // row m = v + 8*lhalf
    }
    __syncthreads();

    // ---------------- Phase 3: gather codebook rows -> NCHW output ---------
    {
        float* outb = out + ((size_t)bidx << 20);
        const int p  = tid & 127;            // position, coalesced over hw
        const int dh = tid >> 7;             // 0/1: which channel of the pair
        const float* crow = cbook + (size_t)bsh[p] * CCH;
        #pragma unroll 4
        for (int i = 0; i < 128; ++i) {
            const int d = i * 2 + dh;
            outb[((size_t)d << 12) + hw0 + p] = crow[d];
        }
    }
}

extern "C" void kernel_launch(void* const* d_in, const int* in_sizes, int n_in,
                              void* d_out, int out_size, void* d_ws, size_t ws_size,
                              hipStream_t stream) {
    const float* x      = (const float*)d_in[0];
    const float* conv_w = (const float*)d_in[1];
    const float* conv_b = (const float*)d_in[2];
    const float* cbook  = (const float*)d_in[3];
    float* out = (float*)d_out;

    hipFuncSetAttribute((const void*)vq_fused_kernel,
                        hipFuncAttributeMaxDynamicSharedMemorySize, SMEM_BYTES);
    vq_fused_kernel<<<NBLK, 256, SMEM_BYTES, stream>>>(x, conv_w, conv_b, cbook, out);
}
